// new_clsnet_75247827026629
// MI455X (gfx1250) — matrix operand order repaired and checked by reading, not compiled, not run on hardware
//
#include <hip/hip_runtime.h>
#include <hip/hip_bf16.h>

typedef __bf16 bf16_t;
typedef __attribute__((ext_vector_type(16))) __bf16 v16bf;
typedef __attribute__((ext_vector_type(8)))  __bf16 v8bf;
typedef __attribute__((ext_vector_type(8)))  float  v8f;

#define TILE_BM 128
#define TILE_BN 128
#define TILE_BK 32
#define LDSS    40   // padded LDS row stride (elements) to spread banks

// CDNA5 async global->LDS copy path (ASYNCcnt), with guaranteed fallback.
#if defined(__gfx1250__) && \
    __has_builtin(__builtin_amdgcn_global_load_async_to_lds_b128) && \
    __has_builtin(__builtin_amdgcn_s_wait_asynccnt)
#define USE_ASYNC_LDS 1
typedef int b128_t __attribute__((vector_size(16)));
typedef __attribute__((address_space(1))) b128_t gb128_t;   // global int4
typedef __attribute__((address_space(3))) b128_t lb128_t;   // LDS int4
#else
#define USE_ASYNC_LDS 0
#endif

// ---------------------------------------------------------------------------
// Generic bf16 WMMA GEMM: C[M,N] = A[M,K] * W[N,K]^T (+ epilogue)
// EPI 0: store f32          EPI 1: store bf16
// EPI 2: +bias, ReLU, bf16  EPI 3: +bias, store f32 only for col < nvalid
// Requires M%128==0, N%128==0, K%32==0. 256 threads = 8 waves (wave32).
// ---------------------------------------------------------------------------
template <int EPI>
__global__ __launch_bounds__(256)
void gemm_bf16_wmma(const bf16_t* __restrict__ A, const bf16_t* __restrict__ W,
                    const float* __restrict__ bias, void* __restrict__ Cout,
                    int M, int N, int K, int ldc, int nvalid)
{
    __shared__ __align__(16) bf16_t sA[2][TILE_BM * LDSS];
    __shared__ __align__(16) bf16_t sB[2][TILE_BN * LDSS];

    const int tid  = threadIdx.x;
    const int lane = tid & 31;
    const int wv   = tid >> 5;
    const int wmi  = wv & 3;      // wave row-slab  (4 x 32 rows)
    const int wni  = wv >> 2;     // wave col-slab  (2 x 64 cols)
    const int l15  = lane & 15;
    const int grp  = lane >> 4;

    const long m0 = (long)blockIdx.x * TILE_BM;
    const long n0 = (long)blockIdx.y * TILE_BN;

    const int lr = tid >> 2;        // 0..63 loader row
    const int lc = (tid & 3) * 8;   // 0,8,16,24 loader col (x8 bf16 = 16B)

    v8f acc[2][4];
#pragma unroll
    for (int mt = 0; mt < 2; ++mt)
#pragma unroll
        for (int nt = 0; nt < 4; ++nt)
            acc[mt][nt] = (v8f){0.f,0.f,0.f,0.f,0.f,0.f,0.f,0.f};

    const int nk = K / TILE_BK;

    auto load_tile = [&](int kt, int buf) {
        const long k0 = (long)kt * TILE_BK;
#pragma unroll
        for (int h = 0; h < 2; ++h) {
            const int r = lr + h * 64;
#if USE_ASYNC_LDS
            // Direct global->LDS DMA, no VGPR landing zone; tracked by ASYNCcnt.
            __builtin_amdgcn_global_load_async_to_lds_b128(
                (gb128_t*)(A + (m0 + r) * (long)K + k0 + lc),
                (lb128_t*)&sA[buf][r * LDSS + lc], 0, 0);
            __builtin_amdgcn_global_load_async_to_lds_b128(
                (gb128_t*)(W + (n0 + r) * (long)K + k0 + lc),
                (lb128_t*)&sB[buf][r * LDSS + lc], 0, 0);
#else
            v8bf av = *(const v8bf*)(A + (m0 + r) * (long)K + k0 + lc);
            v8bf wvv = *(const v8bf*)(W + (n0 + r) * (long)K + k0 + lc);
            *(v8bf*)(&sA[buf][r * LDSS + lc]) = av;
            *(v8bf*)(&sB[buf][r * LDSS + lc]) = wvv;
#endif
        }
    };

    load_tile(0, 0);
#if USE_ASYNC_LDS
    __builtin_amdgcn_s_wait_asynccnt(0);
#endif
    __syncthreads();

    for (int kt = 0; kt < nk; ++kt) {
        const int cur = kt & 1;
        if (kt + 1 < nk) load_tile(kt + 1, cur ^ 1);

        // A fragments: lane l15 = row M, grp selects K sub-block per ISA layout
        v16bf af[2];
#pragma unroll
        for (int mt = 0; mt < 2; ++mt) {
            const bf16_t* p = &sA[cur][(wmi * 32 + mt * 16 + l15) * LDSS];
            v8bf lo = *(const v8bf*)(p + grp * 8);        // K = grp*8 .. +7
            v8bf hi = *(const v8bf*)(p + 16 + grp * 8);   // K = 16+grp*8 .. +7
            af[mt] = __builtin_shufflevector(lo, hi,
                       0,1,2,3,4,5,6,7,8,9,10,11,12,13,14,15);
        }
        // B fragments: lane l15 = col N, grp selects K sub-block (same as A)
        v16bf bfg[4];
#pragma unroll
        for (int nt = 0; nt < 4; ++nt) {
            const bf16_t* p = &sB[cur][(wni * 64 + nt * 16 + l15) * LDSS];
            v8bf lo = *(const v8bf*)(p + grp * 8);
            v8bf hi = *(const v8bf*)(p + 16 + grp * 8);
            bfg[nt] = __builtin_shufflevector(lo, hi,
                       0,1,2,3,4,5,6,7,8,9,10,11,12,13,14,15);
        }
#pragma unroll
        for (int mt = 0; mt < 2; ++mt)
#pragma unroll
            for (int nt = 0; nt < 4; ++nt)
                acc[mt][nt] = __builtin_amdgcn_wmma_f32_16x16x32_bf16(
                    false, af[mt], false, bfg[nt], (short)0, acc[mt][nt],
                    false, false);
#if USE_ASYNC_LDS
        __builtin_amdgcn_s_wait_asynccnt(0);   // next tile fully in LDS
#endif
        __syncthreads();
    }

    // Epilogue: C layout — element r of v8f is row (grp*8 + r), col l15
#pragma unroll
    for (int mt = 0; mt < 2; ++mt) {
#pragma unroll
        for (int nt = 0; nt < 4; ++nt) {
            const long r0 = m0 + wmi * 32 + mt * 16 + grp * 8;
            const long c  = n0 + wni * 64 + nt * 16 + l15;
            v8f a = acc[mt][nt];
#pragma unroll
            for (int r = 0; r < 8; ++r) {
                const long row = r0 + r;
                float v = a[r];
                if (EPI == 0) {
                    ((float*)Cout)[row * ldc + c] = v;
                } else if (EPI == 1) {
                    ((bf16_t*)Cout)[row * ldc + c] = (bf16_t)v;
                } else if (EPI == 2) {
                    v += bias[c];
                    v = v > 0.f ? v : 0.f;
                    ((bf16_t*)Cout)[row * ldc + c] = (bf16_t)v;
                } else {
                    if (c < nvalid)
                        ((float*)Cout)[row * ldc + c] = v + bias[c];
                }
            }
        }
    }
}

// ---------------------------------------------------------------------------
// Elementwise f32 -> bf16 cast with zero-padding tail
// ---------------------------------------------------------------------------
__global__ void cast_pad_kernel(const float* __restrict__ src, bf16_t* __restrict__ dst,
                                long n_src, long n_total)
{
    long i = (long)blockIdx.x * blockDim.x + threadIdx.x;
    const long stride = (long)gridDim.x * blockDim.x;
    for (; i < n_total; i += stride)
        dst[i] = (i < n_src) ? (bf16_t)src[i] : (bf16_t)0.0f;
}

// ---------------------------------------------------------------------------
// Row L2-normalize -> bf16 (and optional raw bf16 copy); pad rows -> zeros
// ---------------------------------------------------------------------------
__global__ void rownorm_cast_kernel(const float* __restrict__ src, bf16_t* __restrict__ nrm,
                                    bf16_t* __restrict__ raw, int nrows, int d)
{
    const int row = blockIdx.x;
    const int tid = threadIdx.x;
    __shared__ float red[256];
    if (row >= nrows) {  // zero-pad rows (uniform branch per block)
        for (int j = tid; j < d; j += 256) nrm[(long)row * d + j] = (bf16_t)0.0f;
        return;
    }
    const float* s = src + (long)row * d;
    float ssq = 0.f;
    for (int j = tid; j < d; j += 256) {
        float v = s[j];
        ssq += v * v;
        if (raw) raw[(long)row * d + j] = (bf16_t)v;
    }
    red[tid] = ssq;
    __syncthreads();
    for (int k = 128; k > 0; k >>= 1) {
        if (tid < k) red[tid] += red[tid + k];
        __syncthreads();
    }
    const float scale = 1.0f / fmaxf(sqrtf(red[0]), 1e-12f);
    for (int j = tid; j < d; j += 256)
        nrm[(long)row * d + j] = (bf16_t)(s[j] * scale);
}

// ---------------------------------------------------------------------------
// Per-row argmax of similarity (== argmin of distance); ties -> lowest index
// ---------------------------------------------------------------------------
__global__ void row_argmax_kernel(const float* __restrict__ sim, int ld, int ncols,
                                  int* __restrict__ idx_out, float* __restrict__ idxf_out)
{
    const int row = blockIdx.x;
    const int tid = threadIdx.x;
    __shared__ float sv[256];
    __shared__ int   si[256];
    const float* s = sim + (long)row * ld;
    float best = -3.402823466e38f;
    int bi = 0;
    for (int j = tid; j < ncols; j += 256) {
        float v = s[j];
        if (v > best) { best = v; bi = j; }
    }
    sv[tid] = best; si[tid] = bi;
    __syncthreads();
    for (int k = 128; k > 0; k >>= 1) {
        if (tid < k) {
            if (sv[tid + k] > sv[tid] ||
                (sv[tid + k] == sv[tid] && si[tid + k] < si[tid])) {
                sv[tid] = sv[tid + k];
                si[tid] = si[tid + k];
            }
        }
        __syncthreads();
    }
    if (tid == 0) { idx_out[row] = si[0]; idxf_out[row] = (float)si[0]; }
}

// ---------------------------------------------------------------------------
// quantized = embeddings[idx] : f32 to output + bf16 for GEMM input
// ---------------------------------------------------------------------------
__global__ void gather_kernel(const float* __restrict__ emb, const int* __restrict__ idx,
                              float* __restrict__ qout, bf16_t* __restrict__ qbf, int d)
{
    const int row = blockIdx.x;
    const int tid = threadIdx.x;
    const float* s = emb + (long)idx[row] * d;
    float* o = qout + (long)row * d;
    bf16_t* ob = qbf + (long)row * d;
    for (int j = tid; j < d; j += 256) {
        float v = s[j];
        o[j] = v;
        ob[j] = (bf16_t)v;
    }
}

// ---------------------------------------------------------------------------
// Per-row 2x2 attention: scores = K^T Q over d, softmax over k, out = V A,
// flat = [out[:,0] | out[:,1]] as bf16.
// KQVx rows: [K(0:d) | Q(d:2d) | V(2d:3d)]
// ---------------------------------------------------------------------------
__global__ void attn_fuse_kernel(const bf16_t* __restrict__ KQVq,
                                 const bf16_t* __restrict__ KQVf,
                                 bf16_t* __restrict__ flat, int d)
{
    const int b = blockIdx.x;
    const int tid = threadIdx.x;
    const bf16_t* rq = KQVq + (long)b * 3 * d;
    const bf16_t* rf = KQVf + (long)b * 3 * d;

    float s00 = 0.f, s01 = 0.f, s10 = 0.f, s11 = 0.f;
    for (int j = tid; j < d; j += 256) {
        float kq = (float)rq[j];
        float kf = (float)rf[j];
        float qq = (float)rq[d + j];
        float qf = (float)rf[d + j];
        s00 += kq * qq; s01 += kq * qf;
        s10 += kf * qq; s11 += kf * qf;
    }
    __shared__ float4 red[256];
    red[tid] = make_float4(s00, s01, s10, s11);
    __syncthreads();
    for (int k = 128; k > 0; k >>= 1) {
        if (tid < k) {
            red[tid].x += red[tid + k].x;
            red[tid].y += red[tid + k].y;
            red[tid].z += red[tid + k].z;
            red[tid].w += red[tid + k].w;
        }
        __syncthreads();
    }
    __shared__ float Amat[4];
    if (tid == 0) {
        const float t00 = red[0].x, t01 = red[0].y, t10 = red[0].z, t11 = red[0].w;
        // softmax over k (axis=1): columns l normalized over {k=0,k=1}
        const float m0 = fmaxf(t00, t10);
        const float e00 = __expf(t00 - m0), e10 = __expf(t10 - m0);
        const float m1 = fmaxf(t01, t11);
        const float e01 = __expf(t01 - m1), e11 = __expf(t11 - m1);
        Amat[0] = e00 / (e00 + e10);  // A[0][0]
        Amat[1] = e01 / (e01 + e11);  // A[0][1]
        Amat[2] = e10 / (e00 + e10);  // A[1][0]
        Amat[3] = e11 / (e01 + e11);  // A[1][1]
    }
    __syncthreads();
    const float a00 = Amat[0], a01 = Amat[1], a10 = Amat[2], a11 = Amat[3];
    bf16_t* out = flat + (long)b * 2 * d;
    for (int j = tid; j < d; j += 256) {
        float vq = (float)rq[2 * d + j];
        float vf = (float)rf[2 * d + j];
        out[j]     = (bf16_t)(vq * a00 + vf * a10);  // token l=0
        out[d + j] = (bf16_t)(vq * a01 + vf * a11);  // token l=1
    }
}

// ---------------------------------------------------------------------------
extern "C" void kernel_launch(void* const* d_in, const int* in_sizes, int n_in,
                              void* d_out, int out_size, void* d_ws, size_t ws_size,
                              hipStream_t stream)
{
    (void)in_sizes; (void)n_in; (void)out_size; (void)ws_size;
    const float* feat   = (const float*)d_in[0];
    const float* emb    = (const float*)d_in[1];
    const float* Kw     = (const float*)d_in[2];
    const float* Qw     = (const float*)d_in[3];
    const float* Vw     = (const float*)d_in[4];
    const float* fuse_w = (const float*)d_in[5];
    const float* fuse_b = (const float*)d_in[6];
    const float* fc_w   = (const float*)d_in[7];
    const float* fc_b   = (const float*)d_in[8];

    const int BS = 8192, D = 2048, NE = 1000, NEP = 1024, NC = 1000, NCP = 1024;

    float* q_out    = (float*)d_out;                  // [BS, D]
    float* pred_out = q_out + (size_t)BS * D;         // [BS, NC]
    float* idxf_out = pred_out + (size_t)BS * NC;     // [BS]

    char* w = (char*)d_ws;
    size_t off = 0;
    auto take = [&](size_t bytes) -> void* {
        off = (off + 255) & ~(size_t)255;
        void* p = w + off;
        off += bytes;
        return p;
    };

    bf16_t* Wkqv  = (bf16_t*)take((size_t)3 * D * D * 2);     // [6144, 2048]
    bf16_t* Wfuse = (bf16_t*)take((size_t)D * 2 * D * 2);     // [2048, 4096]
    bf16_t* Wfc   = (bf16_t*)take((size_t)NCP * D * 2);       // [1024, 2048] padded
    int*    idx_i = (int*)take((size_t)BS * 4);
    bf16_t* Xf    = (bf16_t*)take((size_t)BS * D * 2);        // feat bf16
    bf16_t* Xq    = (bf16_t*)take((size_t)BS * D * 2);        // quantized bf16
    bf16_t* KQVq  = (bf16_t*)take((size_t)BS * 3 * D * 2);    // [BS, 6144]
    bf16_t* KQVf  = (bf16_t*)take((size_t)BS * 3 * D * 2);    // [BS, 6144]

    // Aliases (lifetimes are disjoint; single-stream ordering makes this safe):
    bf16_t* xn   = KQVq;                       // [BS, D]  — dead before KQVq write
    bf16_t* wn   = KQVq + (size_t)BS * D;      // [NEP, D] — dead before KQVq write
    float*  sim  = (float*)KQVf;               // [BS, NEP]— dead before KQVf write
    bf16_t* flat = Xf;                         // [BS, 2D] — covers Xf+Xq after both consumed
    bf16_t* h    = KQVq;                       // [BS, D]  — after KQVq consumed

    const dim3 blk(256);

    // 1) weight casts (Kw|Qw|Vw concatenated row-wise; fc padded to 1024 rows)
    cast_pad_kernel<<<2048, blk, 0, stream>>>(Kw, Wkqv, (long)D * D, (long)D * D);
    cast_pad_kernel<<<2048, blk, 0, stream>>>(Qw, Wkqv + (size_t)D * D, (long)D * D, (long)D * D);
    cast_pad_kernel<<<2048, blk, 0, stream>>>(Vw, Wkqv + (size_t)2 * D * D, (long)D * D, (long)D * D);
    cast_pad_kernel<<<2048, blk, 0, stream>>>(fuse_w, Wfuse, (long)D * 2 * D, (long)D * 2 * D);
    cast_pad_kernel<<<2048, blk, 0, stream>>>(fc_w, Wfc, (long)NC * D, (long)NCP * D);

    // 2) row-normalize (embeddings padded with zero rows; excluded in argmax)
    rownorm_cast_kernel<<<NEP, blk, 0, stream>>>(emb, wn, nullptr, NE, D);
    rownorm_cast_kernel<<<BS, blk, 0, stream>>>(feat, xn, Xf, BS, D);

    // 3) similarity GEMM: sim = xn @ wn^T  (argmin dist == argmax sim)
    gemm_bf16_wmma<0><<<dim3(BS / 128, NEP / 128), blk, 0, stream>>>(
        xn, wn, nullptr, sim, BS, NEP, D, NEP, NEP);

    // 4) argmax + idx output
    row_argmax_kernel<<<BS, blk, 0, stream>>>(sim, NEP, NE, idx_i, idxf_out);

    // 5) gather quantized (fp32 output + bf16 GEMM input)
    gather_kernel<<<BS, blk, 0, stream>>>(emb, idx_i, q_out, Xq, D);

    // 6,7) K/Q/V projections for both tokens: [BS,2048] x [6144,2048]^T
    gemm_bf16_wmma<1><<<dim3(BS / 128, (3 * D) / 128), blk, 0, stream>>>(
        Xq, Wkqv, nullptr, KQVq, BS, 3 * D, D, 3 * D, 3 * D);
    gemm_bf16_wmma<1><<<dim3(BS / 128, (3 * D) / 128), blk, 0, stream>>>(
        Xf, Wkqv, nullptr, KQVf, BS, 3 * D, D, 3 * D, 3 * D);

    // 8) per-row 2x2 attention -> flat [BS, 4096] bf16
    attn_fuse_kernel<<<BS, blk, 0, stream>>>(KQVq, KQVf, flat, D);

    // 9) h = relu(flat @ fuse_w^T + fuse_b) -> bf16
    gemm_bf16_wmma<2><<<dim3(BS / 128, D / 128), blk, 0, stream>>>(
        flat, Wfuse, fuse_b, h, BS, D, 2 * D, D, D);

    // 10) pred = h @ fc_w^T + fc_b (store only 1000 valid cols)
    gemm_bf16_wmma<3><<<dim3(BS / 128, NCP / 128), blk, 0, stream>>>(
        h, Wfc, fc_b, pred_out, BS, NCP, D, NC, NC);
}
